// QLSTM_65481071396784
// MI455X (gfx1250) — compile-verified, matrix operand order checked by reading, not run on hardware
//
#include <hip/hip_runtime.h>
#include <cstdint>

// ---------------------------------------------------------------------------
// QLSTM on gfx1250 (MI455X, wave32).
//  inputs: (T=512, B=512, 256) f32 ; per gate W (264,8), b (8), p (16) f32
//  out   : ys (T,B,8) ++ hT (B,8) ++ cT (B,8)  f32
// ---------------------------------------------------------------------------

typedef __attribute__((ext_vector_type(16))) _Float16 v16h;
typedef __attribute__((ext_vector_type(8)))  float    v8f;
typedef __attribute__((ext_vector_type(4)))  unsigned int v4u;
typedef __attribute__((ext_vector_type(8)))  int      v8i;
typedef __attribute__((ext_vector_type(4)))  int      v4i;

#define TT   512   // timesteps
#define BB   512   // batch
#define IND  256   // input dim
#define NQ   8     // qubits / wires
#define NC   32    // 4 gates * 8 wires (fused gate columns)
#define DKT  264   // IND + NQ

#if __has_builtin(__builtin_amdgcn_tensor_load_to_lds) && \
    __has_builtin(__builtin_amdgcn_s_wait_tensorcnt)
#define QL_USE_TDM 1
#else
#define QL_USE_TDM 0
#endif

__device__ __forceinline__ float sigmoidf_(float x) {
  return 1.0f / (1.0f + __expf(-x));
}

// ---------------------------------------------------------------------------
// Kernel 1: fold DEPTH(=2) and bias/p into packed weights.
//   Bcat (f16, 256 x 32): input-part weights * 2
//   Bh   (f32,   8 x 32): hidden-part weights * 2
//   beta (f32,       32): 2*b + p[0,:] + p[1,:]
// ---------------------------------------------------------------------------
__global__ void qlstm_prep(
    const float* __restrict__ W0, const float* __restrict__ b0, const float* __restrict__ p0,
    const float* __restrict__ W1, const float* __restrict__ b1, const float* __restrict__ p1,
    const float* __restrict__ W2, const float* __restrict__ b2, const float* __restrict__ p2,
    const float* __restrict__ W3, const float* __restrict__ b3, const float* __restrict__ p3,
    _Float16* __restrict__ Bcat, float* __restrict__ Bh, float* __restrict__ beta)
{
  int idx = blockIdx.x * blockDim.x + threadIdx.x;
  if (idx < DKT * NC) {
    int k = idx / NC;
    int c = idx % NC;
    int g = c >> 3, j = c & 7;
    const float* W = (g == 0) ? W0 : (g == 1) ? W1 : (g == 2) ? W2 : W3;
    float w2 = 2.0f * W[k * NQ + j];            // DEPTH = 2 folded in
    if (k < IND) Bcat[k * NC + c] = (_Float16)w2;
    else         Bh[(k - IND) * NC + c] = w2;
  }
  if (idx < NC) {
    int g = idx >> 3, j = idx & 7;
    const float* b = (g == 0) ? b0 : (g == 1) ? b1 : (g == 2) ? b2 : b3;
    const float* p = (g == 0) ? p0 : (g == 1) ? p1 : (g == 2) ? p2 : p3;
    beta[idx] = 2.0f * b[j] + p[j] + p[NQ + j]; // 2b + sum over depth of p
  }
}

// ---------------------------------------------------------------------------
// Kernel 2: Xproj (T*B, 32) = inputs (T*B, 256) @ Bcat (256, 32)
// One wave32 per 16x16 output tile, 8 x v_wmma_f32_16x16x32_f16 per tile.
// Bcat staged into LDS via the Tensor Data Mover (one DMA per block), then
// transposed LDS->LDS to column-major so each lane reads 16B runs.
// A-fragment layout (ISA 7.12.2, 16-bit A 16x32):
//   lane L: row = L&15, kb = (L>>4)*8; VGPRs 0..3 -> K=kb..kb+7,
//   VGPRs 4..7 -> K=16+kb..16+kb+7 (2 f16 per VGPR).
// ---------------------------------------------------------------------------
__global__ void __launch_bounds__(256) qlstm_gemm(
    const float* __restrict__ A,        // (T*B, 256) f32
    const _Float16* __restrict__ Bcat,  // (256, 32) f16, row-major
    float* __restrict__ Xproj)          // (T*B, 32) f32
{
  __shared__ _Float16 sBrow[IND * NC];  // 16 KB row-major staging (TDM target)
  __shared__ _Float16 Blt[NC * IND];    // 16 KB, Blt[c*256 + k] = Bcat[k*32 + c]

#if QL_USE_TDM
  // ---- Tensor Data Mover: one 16 KB 2D DMA (8192 x 1 elements of 2B) ----
  if (threadIdx.x < 32) {               // wave 0 issues the DMA
    unsigned lds = (unsigned)(uintptr_t)&sBrow[0];      // generic lo32 = LDS addr
    unsigned long long ga = (unsigned long long)(uintptr_t)Bcat;
    // D# group 0: count=1 (valid user desc), lds_addr, global_addr, type=2
    v4u g0 = { 1u,
               lds,
               (unsigned)(ga & 0xFFFFFFFFu),
               (unsigned)(((ga >> 32) & 0x01FFFFFFu) | (2u << 30)) };
    // D# group 1: wg_mask=0, data_size=1(2B); tensor_dim0=8192, tensor_dim1=1;
    //             tile_dim0=8192, tile_dim1=1; tensor_dim0_stride=8192
    v8i g1 = { (int)(1u << 16),         // [17:16] data_size = 1 (2 bytes)
               (int)(0x2000u << 16),    // tensor_dim0[15:0] = 8192 (bits 63:48)
               (int)(1u << 16),        // tensor_dim0 hi = 0 | tensor_dim1 lo = 1
               (int)(0x2000u << 16),    // tensor_dim1 hi = 0 | tile_dim0 = 8192
               1,                       // tile_dim1 = 1, tile_dim2 = 0
               8192,                    // tensor_dim0_stride lo
               0, 0 };
    v4i gz = { 0, 0, 0, 0 };
#if __clang_major__ >= 23
    v8i z8 = { 0, 0, 0, 0, 0, 0, 0, 0 };
    __builtin_amdgcn_tensor_load_to_lds(g0, g1, gz, gz, z8, 0);
#else
    __builtin_amdgcn_tensor_load_to_lds(g0, g1, gz, gz, 0);
#endif
    __builtin_amdgcn_s_wait_tensorcnt(0);
  }
  __syncthreads();
  for (int i = threadIdx.x; i < NC * IND; i += 256) {
    int c = i >> 8;                     // column
    int k = i & 255;                    // k index
    Blt[i] = sBrow[k * NC + c];         // LDS->LDS transpose
  }
#else
  for (int i = threadIdx.x; i < NC * IND; i += 256) {
    int c = i >> 8;
    int k = i & 255;
    Blt[i] = Bcat[k * NC + c];
  }
#endif
  __syncthreads();

  const int wave = threadIdx.x >> 5;
  const int lane = threadIdx.x & 31;
  const int tile  = blockIdx.x * 8 + wave;  // 32768 tiles total
  const int tileM = tile >> 1;
  const int tileN = tile & 1;

  const int arow = tileM * 16 + (lane & 15);
  const int kb   = (lane >> 4) * 8;
  const int ncol = tileN * 16 + (lane & 15);

  const float* __restrict__ Arow = A + (size_t)arow * IND;
  const _Float16* __restrict__ Bcol = &Blt[ncol * IND];

  v8f acc = {};
#pragma unroll
  for (int k0 = 0; k0 < IND; k0 += 32) {
    // ---- A fragment: fp32 global loads, convert to f16 in regs ----
    float4 x0 = *(const float4*)(Arow + k0 + kb);
    float4 x1 = *(const float4*)(Arow + k0 + kb + 4);
    float4 x2 = *(const float4*)(Arow + k0 + 16 + kb);
    float4 x3 = *(const float4*)(Arow + k0 + 16 + kb + 4);
    v16h a;
    a[0]  = (_Float16)x0.x; a[1]  = (_Float16)x0.y;
    a[2]  = (_Float16)x0.z; a[3]  = (_Float16)x0.w;
    a[4]  = (_Float16)x1.x; a[5]  = (_Float16)x1.y;
    a[6]  = (_Float16)x1.z; a[7]  = (_Float16)x1.w;
    a[8]  = (_Float16)x2.x; a[9]  = (_Float16)x2.y;
    a[10] = (_Float16)x2.z; a[11] = (_Float16)x2.w;
    a[12] = (_Float16)x3.x; a[13] = (_Float16)x3.y;
    a[14] = (_Float16)x3.z; a[15] = (_Float16)x3.w;

    // ---- B fragment from LDS (mirror layout: lane -> col, K-striped) ----
    const _Float16* bp0 = Bcol + k0 + kb;
    const _Float16* bp1 = Bcol + k0 + 16 + kb;
    v16h b;
#pragma unroll
    for (int j = 0; j < 8; ++j) { b[j] = bp0[j]; b[8 + j] = bp1[j]; }

    acc = __builtin_amdgcn_wmma_f32_16x16x32_f16(
        /*neg_a=*/false, a, /*neg_b=*/false, b,
        /*c_mod=*/(short)0, acc, /*reuse_a=*/false, /*reuse_b=*/false);
  }

  // C layout: lanes 0-15 -> N=lane, M=r ; lanes 16-31 -> N=lane-16, M=8+r
  const int orow = tileM * 16 + (lane >> 4) * 8;
#pragma unroll
  for (int r = 0; r < 8; ++r)
    Xproj[(size_t)(orow + r) * NC + ncol] = acc[r];
}

// ---------------------------------------------------------------------------
// Kernel 3: sequential LSTM recurrence, one wave32 per batch row.
// Per t: phi = Xproj[t,row,:] + h@Bh + beta (replicated per lane);
// lane l evaluates basis states b = l*8 .. l*8+7:
//   ph_b = sum_w phi_w * (bit(b,w)-0.5);  k_b = cos^2 + sin^2;
//   ew_w += k_b * (1 - 2*cumparity_w(b));
// butterfly shfl_xor reduction -> all lanes hold the 8 expectations/gate.
// ---------------------------------------------------------------------------
__global__ void __launch_bounds__(256) qlstm_recur(
    const float* __restrict__ Xproj,  // (T*B, 32)
    const float* __restrict__ Bh,     // (8, 32)
    const float* __restrict__ beta,   // (32)
    float* __restrict__ ys,           // (T, B, 8)
    float* __restrict__ hT,           // (B, 8)
    float* __restrict__ cT)           // (B, 8)
{
  __shared__ float sBh[NQ * NC];
  __shared__ float sbeta[NC];
  if (threadIdx.x < NQ * NC) sBh[threadIdx.x] = Bh[threadIdx.x];
  if (threadIdx.x < NC)      sbeta[threadIdx.x] = beta[threadIdx.x];
  __syncthreads();

  const int wave = threadIdx.x >> 5;
  const int lane = threadIdx.x & 31;
  const int row  = blockIdx.x * 8 + wave;   // 64 blocks * 8 waves = 512 rows

  float h[NQ], c[NQ];
#pragma unroll
  for (int w = 0; w < NQ; ++w) { h[w] = 0.0f; c[w] = 0.0f; }

  for (int t = 0; t < TT; ++t) {
    const float* __restrict__ xp = Xproj + ((size_t)t * BB + row) * NC;
    // CDNA5 speculative prefetch of next timestep's projection row
    __builtin_prefetch(xp + (size_t)BB * NC, 0, 1);

    // phi (replicated in every lane): 32 fused gate columns
    float phi[NC];
#pragma unroll
    for (int cc = 0; cc < NC; ++cc) {
      float s = xp[cc] + sbeta[cc];
#pragma unroll
      for (int w = 0; w < NQ; ++w) s = __fmaf_rn(h[w], sBh[w * NC + cc], s);
      phi[cc] = s;
    }

    float gv[4][NQ];
#pragma unroll
    for (int g = 0; g < 4; ++g) {
      float ew[NQ];
#pragma unroll
      for (int w = 0; w < NQ; ++w) ew[w] = 0.0f;

#pragma unroll
      for (int bi = 0; bi < 8; ++bi) {
        const int bst = lane * 8 + bi;   // basis state 0..255
        float ph = 0.0f;
#pragma unroll
        for (int w = 0; w < NQ; ++w)
          ph = __fmaf_rn(phi[g * NQ + w], ((bst >> w) & 1) ? 0.5f : -0.5f, ph);
        const float cp = __cosf(ph);
        const float sp = __sinf(ph);
        const float kmag = __fmaf_rn(cp, cp, sp * sp);   // |amp|^2 * N_STATES
        int par = 0;
#pragma unroll
        for (int w = 0; w < NQ; ++w) {
          par ^= (bst >> w) & 1;                         // cumulative parity
          ew[w] += par ? -kmag : kmag;                   // Z eigenvalue
        }
      }
      // wave32 butterfly reduction: every lane ends with the full sum
      for (int off = 16; off > 0; off >>= 1) {
#pragma unroll
        for (int w = 0; w < NQ; ++w) ew[w] += __shfl_xor(ew[w], off);
      }
#pragma unroll
      for (int w = 0; w < NQ; ++w) gv[g][w] = ew[w] * (1.0f / 256.0f);
    }

    // LSTM cell update (replicated across lanes; all lanes agree)
#pragma unroll
    for (int w = 0; w < NQ; ++w) {
      const float f  = sigmoidf_(gv[0][w]);
      const float ig = sigmoidf_(gv[1][w]);
      const float gg = tanhf(gv[2][w]);
      const float o  = sigmoidf_(gv[3][w]);
      c[w] = f * c[w] + ig * gg;
      h[w] = o * tanhf(c[w]);
    }
    if (lane < NQ)
      ys[((size_t)t * BB + row) * NQ + lane] = h[lane];
  }

  if (lane < NQ) {
    hT[row * NQ + lane] = h[lane];
    cT[row * NQ + lane] = c[lane];
  }
}

// ---------------------------------------------------------------------------
extern "C" void kernel_launch(void* const* d_in, const int* in_sizes, int n_in,
                              void* d_out, int out_size, void* d_ws, size_t ws_size,
                              hipStream_t stream) {
  (void)in_sizes; (void)n_in; (void)out_size; (void)ws_size;

  const float* inputs = (const float*)d_in[0];
  const float* Wf = (const float*)d_in[1];
  const float* bf = (const float*)d_in[2];
  const float* pf = (const float*)d_in[3];
  const float* Wi = (const float*)d_in[4];
  const float* bi = (const float*)d_in[5];
  const float* pi = (const float*)d_in[6];
  const float* Wu = (const float*)d_in[7];
  const float* bu = (const float*)d_in[8];
  const float* pu = (const float*)d_in[9];
  const float* Wo = (const float*)d_in[10];
  const float* bo = (const float*)d_in[11];
  const float* po = (const float*)d_in[12];

  float* out = (float*)d_out;
  float* ys  = out;                          // (T,B,8)
  float* hT  = out + (size_t)TT * BB * NQ;   // (B,8)
  float* cT  = hT + (size_t)BB * NQ;         // (B,8)

  char* ws = (char*)d_ws;
  _Float16* Bcat = (_Float16*)ws;                    // 16384 B
  float*    Bh   = (float*)(ws + 16384);             // 1024 B
  float*    beta = (float*)(ws + 16384 + 1024);      // 128 B
  float*    Xproj = (float*)(ws + 32768);            // 262144*32*4 = 32 MiB

  // 1) fold weights
  qlstm_prep<<<(DKT * NC + 255) / 256, 256, 0, stream>>>(
      Wf, bf, pf, Wi, bi, pi, Wu, bu, pu, Wo, bo, po, Bcat, Bh, beta);

  // 2) big hoisted GEMM: (T*B,256)@(256,32) -> Xproj
  //    32768 16x16 tiles / 8 waves per block = 4096 blocks
  qlstm_gemm<<<4096, 256, 0, stream>>>(inputs, Bcat, Xproj);

  // 3) sequential recurrence: 512 rows / 8 waves per block = 64 blocks
  qlstm_recur<<<64, 256, 0, stream>>>(Xproj, Bh, beta, ys, hT, cT);
}
